// MoE_TransformerDecoder_63436666962298
// MI455X (gfx1250) — matrix operand order repaired and checked by reading, not compiled, not run on hardware
//
#include <hip/hip_runtime.h>
#include <hip/hip_bf16.h>

// ---------------- model constants ----------------
static constexpr int BB   = 4;
static constexpr int NN   = 1024;
static constexpr int CC   = 1024;
static constexpr int HH   = 16;
static constexpr int DD   = 64;
static constexpr int HID  = 2048;
static constexpr int EE   = 8;
static constexpr long TOK = (long)BB * NN;   // 4096

// ---------------- helpers ----------------
__device__ __forceinline__ unsigned short f2bf(float f) {
    unsigned u = __float_as_uint(f);
    u += 0x7FFFu + ((u >> 16) & 1u);          // round-to-nearest-even
    return (unsigned short)(u >> 16);
}
__device__ __forceinline__ float bf2f(unsigned short h) {
    return __uint_as_float(((unsigned)h) << 16);
}

typedef __bf16 bf16_t;
typedef bf16_t v16bf __attribute__((ext_vector_type(16)));
typedef float  v8f   __attribute__((ext_vector_type(8)));

union FragAB { v16bf v; unsigned u[8]; uint4 q[2]; };

// async 16B global->LDS copy, per lane (ASYNCcnt-tracked)
__device__ __forceinline__ void async_g2l_b128(unsigned lds_byte_off, const void* gaddr) {
    unsigned long long ga = (unsigned long long)gaddr;
    asm volatile("global_load_async_to_lds_b128 %0, %1, off"
                 :: "v"(lds_byte_off), "v"(ga) : "memory");
}
__device__ __forceinline__ void wait_asynccnt0() {
    asm volatile("s_wait_asynccnt 0x0" ::: "memory");
}

// ---------------- f32 -> bf16 convert (grid-stride) ----------------
__global__ void f32_to_bf16_k(const float* __restrict__ s, unsigned short* __restrict__ d, long n) {
    long i = (long)blockIdx.x * blockDim.x + threadIdx.x;
    long stride = (long)gridDim.x * blockDim.x;
    for (; i < n; i += stride) d[i] = f2bf(s[i]);
}

// ---------------- layernorm: f32 row -> bf16 row ----------------
__global__ __launch_bounds__(256) void ln_bf16_k(const float* __restrict__ x,
                                                 const float* __restrict__ g,
                                                 const float* __restrict__ b,
                                                 unsigned short* __restrict__ o) {
    int row = blockIdx.x;
    const float* xr = x + (long)row * CC;
    __shared__ float red[256];
    int tid = threadIdx.x;

    float s = 0.f;
    for (int c = tid; c < CC; c += 256) s += xr[c];
    red[tid] = s; __syncthreads();
    for (int st = 128; st > 0; st >>= 1) { if (tid < st) red[tid] += red[tid + st]; __syncthreads(); }
    float mu = red[0] / (float)CC;
    __syncthreads();

    float v = 0.f;
    for (int c = tid; c < CC; c += 256) { float d0 = xr[c] - mu; v += d0 * d0; }
    red[tid] = v; __syncthreads();
    for (int st = 128; st > 0; st >>= 1) { if (tid < st) red[tid] += red[tid + st]; __syncthreads(); }
    float var = red[0] / (float)CC;
    float rstd = rsqrtf(var + 1e-5f);

    unsigned short* orow = o + (long)row * CC;
    for (int c = tid; c < CC; c += 256)
        orow[c] = f2bf((xr[c] - mu) * rstd * g[c] + b[c]);
}

// ---------------- WMMA bf16 GEMM (double-buffered LDS + async DMA) ----------------
// C[M,N](f32) = A[M,K](bf16) @ W[K,N](bf16) (+bias)
// optional: row gather on A, device row count, residual accumulate,
// per-row-scaled atomic scatter-add (MoE second GEMM).
// A tile staged via GLOBAL_LOAD_ASYNC_TO_LDS_B128 (ASYNCcnt) into the
// *other* LDS buffer while WMMAs run on the current one; one barrier/K-step.
#define FLAG_BIAS   1
#define FLAG_ACC    2
#define FLAG_ATOMIC 4

__global__ __launch_bounds__(256) void gemm_bf16_wmma(
    const unsigned short* __restrict__ A, int lda,
    const unsigned short* __restrict__ W, int ldb,
    const float* __restrict__ bias,
    float* __restrict__ Cp, int ldc,
    int M, int N, int K,
    const int*  __restrict__ rowIdxA,
    const int*  __restrict__ rowIdxOut,
    const float* __restrict__ rowScale,
    const int*  __restrict__ countPtr,
    int flags)
{
    int Meff = countPtr ? *countPtr : M;
    int mBase = blockIdx.y * 128;
    int nBase = blockIdx.x * 64;
    if (mBase >= Meff) return;

    // dynamic LDS, raw byte offsets (base 0) feed the async DMA directly.
    // buffer b: A at [b*12288, +8192), B at [b*12288+8192, +4096)
    extern __shared__ char smem[];

    int tid  = threadIdx.x;
    int lane = tid & 31;
    int wave = tid >> 5;
    int mw = (wave & 3) * 32;   // wave row offset
    int nw = (wave >> 2) * 32;  // wave col offset

    v8f acc[2][2] = {};

    // ---- K-invariant staging addresses ----
    // A tile: 512 x 16B chunks; thread covers chunks tid and tid+256.
    const unsigned short* agsrc[2];
    unsigned aldsoff[2];
    #pragma unroll
    for (int j = 0; j < 2; ++j) {
        int u  = tid + j * 256;      // chunk id
        int r  = u >> 2;             // tile row 0..127
        int c4 = u & 3;              // 16B chunk within row
        int gr = mBase + r;
        int grc = (gr < Meff) ? gr : (Meff - 1);          // clamp: tail rows masked in epilogue
        int ar  = rowIdxA ? rowIdxA[grc] : grc;
        agsrc[j]   = A + (long)ar * lda + c4 * 8;
        aldsoff[j] = (unsigned)(r * 64 + c4 * 16);        // bytes within A region
    }
    // B tile: 1024 uints; thread covers u = tid + i*256, kk = u&31, np = u>>5.
    const unsigned short* bgsrc[4];
    int bnp[4], bkk[4];
    #pragma unroll
    for (int i = 0; i < 4; ++i) {
        int u = tid + i * 256;
        bkk[i] = u & 31;
        bnp[i] = u >> 5;
        bgsrc[i] = W + (long)bkk[i] * ldb + nBase + (bnp[i] << 1);   // N multiple of 64: no guard
    }

    auto stage = [&](int buf, int k0) {
        unsigned base = (unsigned)(buf * 12288);
        #pragma unroll
        for (int j = 0; j < 2; ++j)
            async_g2l_b128(base + aldsoff[j], (const void*)(agsrc[j] + k0));
        unsigned short* Bs = (unsigned short*)(smem + buf * 12288 + 8192);
        #pragma unroll
        for (int i = 0; i < 4; ++i) {
            unsigned val = *(const unsigned*)(bgsrc[i] + (long)k0 * ldb);
            Bs[((bnp[i] << 1) + 0) * 32 + bkk[i]] = (unsigned short)(val & 0xffffu);
            Bs[((bnp[i] << 1) + 1) * 32 + bkk[i]] = (unsigned short)(val >> 16);
        }
        if (k0 + 32 < K)
            __builtin_prefetch((const void*)(W + (long)(k0 + 32) * ldb + nBase), 0, 1);
    };

    int arow  = lane & 15;
    int ahalf = lane >> 4;      // 0/1
    int aofsU = ahalf * 8;      // A frag: K half offset in uints
    int bofsU = ahalf * 8;      // B frag: K half offset in uints

    int nT = K >> 5;            // K multiple of 32
    stage(0, 0);
    wait_asynccnt0();
    __syncthreads();

    for (int t = 0; t < nT; ++t) {
        int buf = t & 1;
        if (t + 1 < nT) stage(buf ^ 1, (t + 1) << 5);   // DMA next tile while computing

        const unsigned*       As = (const unsigned*)(smem + buf * 12288);
        const unsigned short* Bs = (const unsigned short*)(smem + buf * 12288 + 8192);

        // bf16 frag layouts used here (contiguous k for A and B, not the ISA table order):
        // A: lane(0-15)=row M, elems 0-15 -> K=16*ahalf .. +15
        // B: lane(0-15)=col N, elems 0-15 -> K=16*ahalf .. +15
        FragAB afr[2], bfr[2];
        #pragma unroll
        for (int mt = 0; mt < 2; ++mt) {
            const unsigned* ap = &As[(mw + mt * 16 + arow) << 4];
            afr[mt].q[0] = *(const uint4*)(ap + aofsU);        // 16B aligned
            afr[mt].q[1] = *(const uint4*)(ap + 4 + aofsU);
        }
        #pragma unroll
        for (int nt = 0; nt < 2; ++nt) {
            const uint4* bp4 = (const uint4*)((const unsigned*)&Bs[(nw + nt * 16 + arow) * 32] + bofsU);
            bfr[nt].q[0] = bp4[0];
            bfr[nt].q[1] = bp4[1];
        }
        #pragma unroll
        for (int mt = 0; mt < 2; ++mt)
            #pragma unroll
            for (int nt = 0; nt < 2; ++nt)
                acc[mt][nt] = __builtin_amdgcn_wmma_f32_16x16x32_bf16(
                    false, afr[mt].v, false, bfr[nt].v, (short)0, acc[mt][nt], false, false);

        wait_asynccnt0();      // own DMA for next tile done (and frag reads consumed by WMMA)
        __syncthreads();       // single barrier per K-step
    }

    // epilogue: C/D layout — lane(0-15)=col, VGPR j -> row j + 8*ahalf
    #pragma unroll
    for (int mt = 0; mt < 2; ++mt) {
        #pragma unroll
        for (int nt = 0; nt < 2; ++nt) {
            #pragma unroll
            for (int j = 0; j < 8; ++j) {
                int m = mBase + mw + mt * 16 + (ahalf << 3) + j;
                int n = nBase + nw + nt * 16 + arow;
                if (m < Meff && n < N) {
                    float v = acc[mt][nt][j];
                    if (flags & FLAG_BIAS) v += bias[n];
                    if (flags & FLAG_ATOMIC) {
                        atomicAdd(Cp + (long)rowIdxOut[m] * ldc + n, v * rowScale[m]);
                    } else if (flags & FLAG_ACC) {
                        Cp[(long)m * ldc + n] += v;
                    } else {
                        Cp[(long)m * ldc + n] = v;
                    }
                }
            }
        }
    }
}

// ---------------- attention: one wave per (b,h,q), online softmax ----------------
__global__ __launch_bounds__(32) void attn_k(const float* __restrict__ qkv, float* __restrict__ out) {
    int lane = threadIdx.x;
    int idx = blockIdx.x;            // (b*H + h)*N + q
    int q  = idx & (NN - 1);
    int bh = idx >> 10;
    int h  = bh & (HH - 1);
    int b  = bh >> 4;
    const float scale = 0.125f;      // 1/sqrt(64)

    long base = (long)b * NN * 3 * CC;
    long qoff = base + ((long)q * 3) * CC + h * DD;
    float q0 = qkv[qoff + lane]      * scale;
    float q1 = qkv[qoff + 32 + lane] * scale;

    float m = -3.0e38f, l = 0.f, a0 = 0.f, a1 = 0.f;
    for (int k = 0; k < NN; ++k) {
        long kb = base + ((long)k * 3) * CC + h * DD;
        float s = q0 * qkv[kb + CC + lane] + q1 * qkv[kb + CC + 32 + lane];
        #pragma unroll
        for (int o = 16; o > 0; o >>= 1) s += __shfl_xor(s, o, 32);
        float mn = fmaxf(m, s);
        float f  = __expf(m - mn);
        float p  = __expf(s - mn);
        l = l * f + p;
        a0 = a0 * f + p * qkv[kb + 2 * CC + lane];
        a1 = a1 * f + p * qkv[kb + 2 * CC + 32 + lane];
        m = mn;
    }
    float inv = 1.f / l;
    long ooff = ((long)(b * NN + q)) * CC + h * DD;
    out[ooff + lane]      = a0 * inv;
    out[ooff + 32 + lane] = a1 * inv;
}

// ---------------- elementwise ----------------
__global__ void gelu_bf16_k(const float* __restrict__ in, unsigned short* __restrict__ out, long n) {
    long i = (long)blockIdx.x * blockDim.x + threadIdx.x;
    if (i >= n) return;
    float v = in[i];
    out[i] = f2bf(0.5f * v * (1.f + erff(v * 0.70710678118f)));
}

__global__ __launch_bounds__(256) void silu_mul_bf16_k(const float* __restrict__ t1,
                                                       const float* __restrict__ t3,
                                                       unsigned short* __restrict__ h,
                                                       const int* __restrict__ cntp) {
    long limit = (long)(*cntp) * HID;
    long i = (long)blockIdx.x * 256 + threadIdx.x;
    if (i >= limit) return;
    float a = t1[i];
    float s = a / (1.f + expf(-a));
    h[i] = f2bf(s * t3[i]);
}

__global__ void zero_i32_k(int* p, int n) {
    int i = blockIdx.x * blockDim.x + threadIdx.x;
    if (i < n) p[i] = 0;
}

// ---------------- MoE routing: gate scores + softmax + top-2, build lists ----------------
__global__ __launch_bounds__(256) void moe_route_k(const unsigned short* __restrict__ ln,
                                                   const float* __restrict__ gate_w,
                                                   int* __restrict__ cnt,
                                                   int* __restrict__ idxl,
                                                   float* __restrict__ wls) {
    int lane = threadIdx.x & 31;
    int wave = threadIdx.x >> 5;
    int token = blockIdx.x * 8 + wave;

    float acc[EE];
    #pragma unroll
    for (int e = 0; e < EE; ++e) acc[e] = 0.f;
    for (int c = lane; c < CC; c += 32) {
        float xv = bf2f(ln[(long)token * CC + c]);
        #pragma unroll
        for (int e = 0; e < EE; ++e) acc[e] += xv * gate_w[e * CC + c];
    }
    #pragma unroll
    for (int e = 0; e < EE; ++e)
        #pragma unroll
        for (int o = 16; o > 0; o >>= 1) acc[e] += __shfl_xor(acc[e], o, 32);

    if (lane == 0) {
        float mx = acc[0];
        #pragma unroll
        for (int e = 1; e < EE; ++e) mx = fmaxf(mx, acc[e]);
        float w[EE], s = 0.f;
        #pragma unroll
        for (int e = 0; e < EE; ++e) { w[e] = expf(acc[e] - mx); s += w[e]; }
        float inv = 1.f / s;
        #pragma unroll
        for (int e = 0; e < EE; ++e) w[e] *= inv;
        int i1 = 0;
        #pragma unroll
        for (int e = 1; e < EE; ++e) if (w[e] > w[i1]) i1 = e;   // ties -> lowest index (top_k order)
        int i2 = (i1 == 0) ? 1 : 0;
        #pragma unroll
        for (int e = 0; e < EE; ++e) if (e != i1 && w[e] > w[i2]) i2 = e;

        int p1 = atomicAdd(&cnt[i1], 1);
        idxl[i1 * (int)TOK + p1] = token;
        wls [i1 * (int)TOK + p1] = w[i1];
        int p2 = atomicAdd(&cnt[i2], 1);
        idxl[i2 * (int)TOK + p2] = token;
        wls [i2 * (int)TOK + p2] = w[i2];
    }
}

// ---------------- host driver ----------------
extern "C" void kernel_launch(void* const* d_in, const int* in_sizes, int n_in,
                              void* d_out, int out_size, void* d_ws, size_t ws_size,
                              hipStream_t stream) {
    (void)in_sizes; (void)n_in; (void)out_size; (void)ws_size;

    // --- workspace bump allocator ---
    char* wsp = (char*)d_ws;
    auto alloc = [&](size_t bytes) -> void* {
        void* p = (void*)wsp;
        wsp += (bytes + 255) & ~(size_t)255;
        return p;
    };
    float*          xres = (float*)         alloc(TOK * CC * 4);        // residual stream
    unsigned short* lnb  = (unsigned short*)alloc(TOK * CC * 2);        // LN output (bf16)
    float*          qkvb = (float*)         alloc(TOK * 3 * CC * 4);    // qkv
    float*          attb = (float*)         alloc(TOK * CC * 4);        // attention out f32
    unsigned short* abf  = (unsigned short*)alloc(TOK * CC * 2);        // attention out bf16
    float*          t1   = (float*)         alloc(TOK * HID * 4);
    float*          t3   = (float*)         alloc(TOK * HID * 4);
    unsigned short* hb   = (unsigned short*)alloc(TOK * HID * 2);
    unsigned short* wbuf = (unsigned short*)alloc((size_t)CC * 3 * CC * 2); // weight staging (max qkv_w)
    int*            cnt  = (int*)           alloc(EE * 4);
    int*            idxl = (int*)           alloc((size_t)EE * TOK * 4);
    float*          wls  = (float*)         alloc((size_t)EE * TOK * 4);

    auto cvt = [&](const float* src, unsigned short* dst, long n) {
        f32_to_bf16_k<<<2048, 256, 0, stream>>>(src, dst, n);
    };
    auto gemm = [&](const unsigned short* A, int lda, const float* Wf32, long wElems, int ldb,
                    const float* bias, float* Cp, int ldc, int M, int N, int K,
                    const int* rIdxA, const int* rIdxO, const float* rScale,
                    const int* cntp, int flags) {
        cvt(Wf32, wbuf, wElems);
        dim3 g((N + 63) / 64, (M + 127) / 128);
        gemm_bf16_wmma<<<g, 256, 24576, stream>>>(A, lda, wbuf, ldb, bias, Cp, ldc, M, N, K,
                                                  rIdxA, rIdxO, rScale, cntp, flags);
    };

    // --- inputs (setup_inputs dict order) ---
    const float* x_in = (const float*)d_in[0];
    int ii = 1;

    hipMemcpyAsync(xres, x_in, TOK * CC * 4, hipMemcpyDeviceToDevice, stream);

    for (int l = 0; l < 4; ++l) {
        const float* ln1g  = (const float*)d_in[ii++];
        const float* ln1b_ = (const float*)d_in[ii++];
        const float* qkv_w = (const float*)d_in[ii++];
        const float* out_w = (const float*)d_in[ii++];
        const float* out_b = (const float*)d_in[ii++];
        const float* ln2g  = (const float*)d_in[ii++];
        const float* ln2b_ = (const float*)d_in[ii++];

        // attention block
        ln_bf16_k<<<(int)TOK, 256, 0, stream>>>(xres, ln1g, ln1b_, lnb);
        gemm(lnb, CC, qkv_w, (long)CC * 3 * CC, 3 * CC, nullptr, qkvb, 3 * CC,
             (int)TOK, 3 * CC, CC, nullptr, nullptr, nullptr, nullptr, 0);
        attn_k<<<BB * HH * NN, 32, 0, stream>>>(qkvb, attb);
        cvt(attb, abf, TOK * CC);
        gemm(abf, CC, out_w, (long)CC * CC, CC, out_b, xres, CC,
             (int)TOK, CC, CC, nullptr, nullptr, nullptr, nullptr, FLAG_BIAS | FLAG_ACC);

        // FFN / MoE block
        ln_bf16_k<<<(int)TOK, 256, 0, stream>>>(xres, ln2g, ln2b_, lnb);

        if (l == 0) {
            const float* ffw1 = (const float*)d_in[ii++];
            const float* ffb1 = (const float*)d_in[ii++];
            const float* ffw2 = (const float*)d_in[ii++];
            const float* ffb2 = (const float*)d_in[ii++];
            gemm(lnb, CC, ffw1, (long)CC * HID, HID, ffb1, t1, HID,
                 (int)TOK, HID, CC, nullptr, nullptr, nullptr, nullptr, FLAG_BIAS);
            gelu_bf16_k<<<(int)((TOK * HID + 255) / 256), 256, 0, stream>>>(t1, hb, TOK * HID);
            gemm(hb, HID, ffw2, (long)HID * CC, CC, ffb2, xres, CC,
                 (int)TOK, CC, HID, nullptr, nullptr, nullptr, nullptr, FLAG_BIAS | FLAG_ACC);
        } else {
            const float* gate_w = (const float*)d_in[ii++];
            const float* exw1   = (const float*)d_in[ii++];
            const float* exb1   = (const float*)d_in[ii++];
            const float* exw2   = (const float*)d_in[ii++];
            const float* exb2   = (const float*)d_in[ii++];
            const float* exw3   = (const float*)d_in[ii++];
            const float* exb3   = (const float*)d_in[ii++];

            zero_i32_k<<<1, EE, 0, stream>>>(cnt, EE);
            moe_route_k<<<(int)(TOK / 8), 256, 0, stream>>>(lnb, gate_w, cnt, idxl, wls);

            for (int e = 0; e < EE; ++e) {
                const int*   il = idxl + (long)e * TOK;
                const float* wl = wls  + (long)e * TOK;
                // t1 = gather(ln) @ w1[e] + b1[e]   (compact rows 0..cnt[e])
                gemm(lnb, CC, exw1 + (long)e * CC * HID, (long)CC * HID, HID,
                     exb1 + (long)e * HID, t1, HID, (int)TOK, HID, CC,
                     il, nullptr, nullptr, cnt + e, FLAG_BIAS);
                // t3 = gather(ln) @ w3[e] + b3[e]
                gemm(lnb, CC, exw3 + (long)e * CC * HID, (long)CC * HID, HID,
                     exb3 + (long)e * HID, t3, HID, (int)TOK, HID, CC,
                     il, nullptr, nullptr, cnt + e, FLAG_BIAS);
                // h = silu(t1) * t3  (bf16, compact)
                silu_mul_bf16_k<<<(int)((TOK * HID + 255) / 256), 256, 0, stream>>>(t1, t3, hb, cnt + e);
                // xres[token] += we * (h @ w2[e] + b2[e])   (atomic scatter)
                gemm(hb, HID, exw2 + (long)e * HID * CC, (long)HID * CC, CC,
                     exb2 + (long)e * CC, xres, CC, (int)TOK, CC, HID,
                     nullptr, il, wl, cnt + e, FLAG_BIAS | FLAG_ATOMIC);
            }
        }
    }

    hipMemcpyAsync(d_out, xres, TOK * CC * 4, hipMemcpyDeviceToDevice, stream);
}
